// YatConv_4346506903795
// MI455X (gfx1250) — matrix operand order repaired and checked by reading, not compiled, not run on hardware
//
#include <hip/hip_runtime.h>
#include <hip/hip_bf16.h>

#define EPSI (1.0f / 137.0f)

typedef __attribute__((ext_vector_type(16))) __bf16 v16bf;
typedef __attribute__((ext_vector_type(8)))  float  v8f;

union Frag {
    uint4  u[2];
    v16bf  v;
};

static __device__ __forceinline__ unsigned short f2bf(float f) {
    // round-to-nearest-even float -> bf16
    unsigned u = __float_as_uint(f);
    unsigned r = u + 0x7FFFu + ((u >> 16) & 1u);
    return (unsigned short)(r >> 16);
}

// ---------------------------------------------------------------------------
// Pre-pass 1: kernel [3,3,128,256] f32 (HWIO, Cout fastest) -> wT bf16 [256][1152]
// wT[n*1152 + kk] with kk = (kh*3+kw)*128 + cin  (== flat f32 index / 256)
// ---------------------------------------------------------------------------
__global__ __launch_bounds__(256) void yat_convert_w(const float* __restrict__ k,
                                                     unsigned short* __restrict__ wT) {
    int t  = blockIdx.x * 256 + threadIdx.x;         // 0 .. 294911
    int n  = t / 1152;
    int kk = t - n * 1152;
    wT[t] = f2bf(k[kk * 256 + n]);                   // coalesced 2B writes
}

// ---------------------------------------------------------------------------
// Pre-pass 2: kernel_sq[n] = sum over 1152 of w^2
// ---------------------------------------------------------------------------
__global__ __launch_bounds__(256) void yat_ksq(const float* __restrict__ k,
                                               float* __restrict__ ksq) {
    int n = threadIdx.x;                             // 0..255
    float s = 0.f;
    for (int kk = 0; kk < 1152; ++kk) {
        float v = k[kk * 256 + n];
        s += v * v;
    }
    ksq[n] = s;
}

// ---------------------------------------------------------------------------
// Main: one block per (b, h-pair). 256 threads = 8 waves.
// Block computes out[b, 2*h2 .. 2*h2+1, 0..63, 0..255]  (M = 128, N = 256).
// Wave wv owns N-tiles {wv*32, wv*32+16} for all 8 M-tiles (halves L2 weight
// traffic vs one-row blocks: 1024 blocks x 590KB = 0.6 GB from L2).
// ---------------------------------------------------------------------------
__global__ __launch_bounds__(256) void yat_main(const float* __restrict__ x,
                                                const unsigned short* __restrict__ wT,
                                                const float* __restrict__ ksq,
                                                const float* __restrict__ bias,
                                                const float* __restrict__ alpha,
                                                float* __restrict__ out) {
    // dynamic LDS: 4x66x128 bf16 slab (67584 B) + colsq 4*66 f32 + psq 128 f32
    extern __shared__ unsigned char smem[];
    unsigned short* xt    = (unsigned short*)smem;           // [4][66][128] bf16
    float*          colsq = (float*)(smem + 4 * 66 * 128 * 2);   // [4*66]
    float*          psq   = colsq + 4 * 66;                      // [128]

    const int tid = threadIdx.x;
    const int bg  = blockIdx.x;                      // b*32 + h2
    const int b   = bg >> 5;
    const int h0  = (bg & 31) * 2;                   // first of 2 output rows

    // ---- zero reductions + halo columns -----------------------------------
    colsq[tid & 255] = 0.0f;                         // tid 0..255 covers 0..255
    if (tid < 8) colsq[256 + tid] = 0.0f;            // remaining 264-256
    {
        // zero pad columns w=-1 and w=64 for all 4 slab rows: 4*2*128 bf16
        int c4   = tid & 31;
        int col2 = (tid >> 5) & 1;
        int r    = tid >> 6;                         // 0..3
        int off  = (r * 66 + (col2 ? 65 : 0)) * 128 + c4 * 4;
        *(uint2*)&xt[off] = make_uint2(0u, 0u);
    }
    __syncthreads();

    // ---- stage x (4 rows h0-1..h0+2, f32 -> bf16) + column squared norms ---
    const float4* xv = (const float4*)x;
    for (int i = tid; i < 4 * 64 * 32; i += 256) {
        int c4 = i & 31;
        int w  = (i >> 5) & 63;
        int r  = i >> 11;                            // 0..3
        int hh = h0 + r - 1;
        float4 v = make_float4(0.f, 0.f, 0.f, 0.f);
        if (hh >= 0 && hh < 64)
            v = xv[((b * 64 + hh) * 64 + w) * 32 + c4];
        unsigned u0 = (unsigned)f2bf(v.x) | ((unsigned)f2bf(v.y) << 16);
        unsigned u1 = (unsigned)f2bf(v.z) | ((unsigned)f2bf(v.w) << 16);
        int off = (r * 66 + w + 1) * 128 + c4 * 4;
        *(uint2*)&xt[off] = make_uint2(u0, u1);
        atomicAdd(&colsq[r * 66 + w + 1],
                  v.x * v.x + v.y * v.y + v.z * v.z + v.w * v.w);  // ds_add_f32
    }
    __syncthreads();

    if (tid < 128) {                                 // ||patch||^2 per (hrow, w)
        int hr = tid >> 6;                           // output row in pair
        int w  = tid & 63;
        float s = 0.f;
#pragma unroll
        for (int r = 0; r < 3; ++r)
#pragma unroll
            for (int dw = 0; dw < 3; ++dw)
                s += colsq[(hr + r) * 66 + w + dw];
        psq[tid] = s;
    }
    __syncthreads();

    // ---- implicit GEMM: M=128 x N=256 x K=1152, bf16 WMMA ------------------
    const int wv     = tid >> 5;
    const int lane   = tid & 31;
    const int half   = lane >> 4;                    // 0: lanes 0-15, 1: lanes 16-31
    const int l16    = lane & 15;
    const int n_base = wv * 32;

    v8f acc[8][2];
#pragma unroll
    for (int mt = 0; mt < 8; ++mt)
#pragma unroll
        for (int nt = 0; nt < 2; ++nt)
            acc[mt][nt] = (v8f){0.f, 0.f, 0.f, 0.f, 0.f, 0.f, 0.f, 0.f};

    for (int ks = 0; ks < 36; ++ks) {                // K chunks of 32: (kh,kw,cin/32)
        int kh = ks / 12;
        int kw = (ks >> 2) % 3;
        int cb = (ks & 3) * 32;

        if (ks + 1 < 36)                             // near-cache weight prefetch
            __builtin_prefetch(&wT[(n_base + l16) * 1152 + (ks + 1) * 32], 0, 3);

        // B fragments (32x16 bf16) from global (L2-resident).
        // Layout: lanes 0-15 -> K 0..15, lanes 16-31 -> K 16..31 (contiguous)
        Frag bfr[2];
#pragma unroll
        for (int nt = 0; nt < 2; ++nt) {
            int n = n_base + nt * 16 + l16;
            const uint4* p = (const uint4*)&wT[n * 1152 + ks * 32 + half * 16];
            bfr[nt].u[0] = p[0];                     // global_load_b128
            bfr[nt].u[1] = p[1];
        }

        // A fragments (16x32 bf16) from LDS, one M-tile at a time.
        // Layout: lanes 0-15 -> K 0..15, lanes 16-31 -> K 16..31 (contiguous)
#pragma unroll
        for (int mt = 0; mt < 8; ++mt) {
            int hr  = mt >> 2;                       // slab row base of this M-tile
            int w   = (mt & 3) * 16 + l16;           // spatial column
            int off = ((hr + kh) * 66 + w + kw) * 128 + cb + half * 16;
            const uint4* p = (const uint4*)&xt[off];
            Frag a;
            a.u[0] = p[0];                           // ds_load_b128
            a.u[1] = p[1];                           // +8 bf16 elements
            acc[mt][0] = __builtin_amdgcn_wmma_f32_16x16x32_bf16(
                false, a.v, false, bfr[0].v, (short)0, acc[mt][0], false, false);
            acc[mt][1] = __builtin_amdgcn_wmma_f32_16x16x32_bf16(
                false, a.v, false, bfr[1].v, (short)0, acc[mt][1], false, false);
        }
    }

    // ---- fused Yat epilogue ------------------------------------------------
    const float al    = alpha[0];
    const float scale = __powf(16.0f / __logf(257.0f), al);  // (sqrt(256)/log1p(256))^a

#pragma unroll
    for (int nt = 0; nt < 2; ++nt) {
        int   n  = n_base + nt * 16 + l16;
        float kq = ksq[n];
        float bs = bias[n];
#pragma unroll
        for (int mt = 0; mt < 8; ++mt) {
#pragma unroll
            for (int r = 0; r < 8; ++r) {
                // C/D layout: VGPR r -> M=r (lanes 0-15), M=r+8 (lanes 16-31)
                int   m    = mt * 16 + r + half * 8; // 0..127
                int   hrow = h0 + (m >> 6);
                int   w    = m & 63;
                float dot  = acc[mt][nt][r];
                float dist = psq[m] + kq - 2.0f * dot;
                float y    = (dot * dot) / (dist + EPSI) * scale + bs;
                out[(((size_t)b * 64 + hrow) * 64 + w) * 256 + n] = y;
            }
        }
    }
}

// ---------------------------------------------------------------------------
extern "C" void kernel_launch(void* const* d_in, const int* in_sizes, int n_in,
                              void* d_out, int out_size, void* d_ws, size_t ws_size,
                              hipStream_t stream) {
    const float* x     = (const float*)d_in[0];   // [32,64,64,128]
    const float* kern  = (const float*)d_in[1];   // [3,3,128,256]
    const float* bias  = (const float*)d_in[2];   // [256]
    const float* alpha = (const float*)d_in[3];   // [1]
    float*       out   = (float*)d_out;           // [32,64,64,256]

    unsigned short* wT  = (unsigned short*)d_ws;                       // 589,824 B
    float*          ksq = (float*)((char*)d_ws + (size_t)256 * 1152 * 2);

    const size_t lds_bytes = (size_t)4 * 66 * 128 * 2   // bf16 slab
                           + (size_t)4 * 66 * 4         // colsq
                           + (size_t)128 * 4;           // psq

    yat_convert_w<<<1152, 256, 0, stream>>>(kern, wT);
    yat_ksq<<<1, 256, 0, stream>>>(kern, ksq);
    yat_main<<<32 * 32, 256, lds_bytes, stream>>>(x, wT, ksq, bias, alpha, out);
}
